// Struct_Attention_63737314673143
// MI455X (gfx1250) — matrix operand order repaired and checked by reading, not compiled, not run on hardware
//
#include <hip/hip_runtime.h>
#include <cstdint>
#include <cstddef>

// ---------------------------------------------------------------------------
// Problem constants (from reference setup_inputs)
// ---------------------------------------------------------------------------
constexpr int B   = 8;
constexpr int S   = 64;
constexpr int H   = 512;
constexpr int T   = 17;
constexpr int NSEQ = B * S;          // 512 sequences per LSTM pass
constexpr int KDIM = 2 * H;          // 1024 = [x | h] fused K
constexpr float NEGV = -1.0e9f;
constexpr float EPSV = 1.0e-5f;

typedef _Float16 half_t;
typedef __attribute__((ext_vector_type(16))) _Float16 v16h;
typedef __attribute__((ext_vector_type(8)))  _Float16 v8h;
typedef __attribute__((ext_vector_type(8)))  float    v8f;

union HF16 { v16h v; v8h h[2]; };

static __device__ __forceinline__ v16h ld_frag(const half_t* p0, const half_t* p1) {
  HF16 u;
  u.h[0] = *(const v8h*)p0;
  u.h[1] = *(const v8h*)p1;
  return u.v;
}

static __device__ __forceinline__ float sigmoidf(float x) {
  return 1.0f / (1.0f + __expf(-x));
}

// gfx1250 async global->LDS copy (16B per lane), tracked by ASYNCcnt.
static __device__ __forceinline__ void async_copy16(uint32_t lds_off, const void* gptr) {
  asm volatile("global_load_async_to_lds_b128 %0, %1, off"
               :: "v"(lds_off), "v"((uint64_t)(uintptr_t)gptr)
               : "memory");
}
static __device__ __forceinline__ void wait_async0() {
  asm volatile("s_wait_asynccnt 0x0" ::: "memory");
}

struct PosV { int ip, jp, valid; };

// mode: 0=rows_f  1=rows_b  2=cols_f  3=cols_b
static __device__ __forceinline__ PosV map_pos(int mode, int rr, int t, int len) {
  PosV p;
  if (mode == 0) {            // rows_f: len = S - i, forward from diagonal
    int j = rr + t;
    p.valid = (j < S); p.ip = rr; p.jp = p.valid ? j : (S - 1);
  } else if (mode == 1) {     // rows_b: len = lengths[b], backward dir
    int j = len - 1 - t;
    p.valid = (t < len); p.ip = rr; p.jp = p.valid ? j : 0;
  } else if (mode == 2) {     // cols_f: len = S, forward over i
    p.valid = 1; p.ip = t; p.jp = rr;
  } else {                    // cols_b: len = j+1, backward over i
    int i = rr - t;
    p.valid = (i >= 0); p.ip = p.valid ? i : 0; p.jp = rr;
  }
  return p;
}

// ---------------------------------------------------------------------------
// Prep kernels
// ---------------------------------------------------------------------------
__global__ void zero_u32(uint32_t* p, size_t n) {
  size_t i = (size_t)blockIdx.x * blockDim.x + threadIdx.x;
  size_t stride = (size_t)gridDim.x * blockDim.x;
  for (; i < n; i += stride) p[i] = 0u;
}

// struct_vec (B, S(S+1)/2, H) -> pv_h (B,S,S,H) f16, zeros below diagonal
__global__ void scatter_pv(const float* __restrict__ sv, half_t* __restrict__ pv) {
  size_t n = (size_t)B * S * S * H;
  size_t i = (size_t)blockIdx.x * blockDim.x + threadIdx.x;
  size_t stride = (size_t)gridDim.x * blockDim.x;
  for (; i < n; i += stride) {
    int d  = (int)(i & (H - 1));
    int j  = (int)((i >> 9) & (S - 1));
    int ii = (int)((i >> 15) & (S - 1));
    int b  = (int)(i >> 21);
    float v = 0.0f;
    if (j >= ii) {
      int off = ii * S - (ii * (ii - 1)) / 2;   // triu row offset
      int l = off + (j - ii);
      v = sv[((size_t)b * (S * (S + 1) / 2) + l) * H + d];
    }
    pv[i] = (half_t)v;
  }
}

// Pack [Wih | Whh] -> f16  Wp[dir][gate][n][k], n in [0,H), k in [0,KDIM)
__global__ void pack_w(const float* __restrict__ Wih_f, const float* __restrict__ Whh_f,
                       const float* __restrict__ Wih_b, const float* __restrict__ Whh_b,
                       half_t* __restrict__ Wp) {
  size_t n = (size_t)2 * 4 * H * KDIM;
  size_t i = (size_t)blockIdx.x * blockDim.x + threadIdx.x;
  size_t stride = (size_t)gridDim.x * blockDim.x;
  for (; i < n; i += stride) {
    int k    = (int)(i & (KDIM - 1));
    int nn   = (int)((i >> 10) & (H - 1));
    int gate = (int)((i >> 19) & 3);
    int dir  = (int)(i >> 21);
    const float* Wih = dir ? Wih_b : Wih_f;
    const float* Whh = dir ? Whh_b : Whh_f;
    size_t row = (size_t)gate * H + nn;
    float v = (k < H) ? Wih[row * H + k] : Whh[row * H + (k - H)];
    Wp[i] = (half_t)v;
  }
}

__global__ void pack_b(const float* __restrict__ bih_f, const float* __restrict__ bhh_f,
                       const float* __restrict__ bih_b, const float* __restrict__ bhh_b,
                       float* __restrict__ bsum) {
  int i = blockIdx.x * blockDim.x + threadIdx.x;
  if (i >= 2 * 4 * H) return;
  int idx = i & (4 * H - 1);
  int dir = i >> 11;
  bsum[i] = dir ? (bih_b[idx] + bhh_b[idx]) : (bih_f[idx] + bhh_f[idx]);
}

__global__ void init_pt(float* __restrict__ pt, const float* __restrict__ bt) {
  size_t n = (size_t)B * T * S * S;
  size_t i = (size_t)blockIdx.x * blockDim.x + threadIdx.x;
  size_t stride = (size_t)gridDim.x * blockDim.x;
  for (; i < n; i += stride) {
    int t = (int)((i >> 12) % T);
    pt[i] = bt[t];
  }
}

// ---------------------------------------------------------------------------
// LSTM timestep kernel: one launch per (pass, t).
// Block = 128 threads (4 waves). Block owns 16 sequences x 64 hidden cols.
// Whole A panel ([x_t | h_prev] : 16 x 1024 f16 = 32KB) is staged into LDS once
// via gfx1250 async global->LDS b128 copies, then the K loop is barrier-free:
// ds_load_b128 fragments + L2-resident weight fragments + v_wmma_f32_16x16x32_f16.
// ---------------------------------------------------------------------------
__global__ void __launch_bounds__(128)
lstm_step(const half_t* __restrict__ pv_h,     // [B][S][S][H] f16
          const half_t* __restrict__ Wp,       // [4][H][KDIM] f16 (this dir)
          const float*  __restrict__ bsum,     // [4][H]       (this dir)
          const half_t* __restrict__ h_in,     // [NSEQ][H] f16
          half_t*       __restrict__ h_out,    // [NSEQ][H] f16
          float*        __restrict__ c_state,  // [NSEQ][H] f32
          const int*    __restrict__ lengths,  // [B]
          const float*  __restrict__ Wt,       // [T][2H] f32
          float*        __restrict__ pt,       // [B][T][S][S]
          int t, int mode, int dimOff) {
  __shared__ __align__(16) half_t Ablock[16][KDIM];  // 16 rows x 1024 K (32KB)
  __shared__ float hy[16][64];                       // y = valid ? h_new : 0

  const int tid  = threadIdx.x;
  const int lane = tid & 31;
  const int wid  = tid >> 5;

  const int rt = blockIdx.x & 31;        // 32 row tiles of 16 seqs
  const int ct = blockIdx.x >> 5;        // 8 col tiles of 64 hidden
  const int rowBase  = rt * 16;
  const int colBase0 = ct * 64;
  const int colBaseW = colBase0 + wid * 16;

  // ---- stage the whole A panel asynchronously: 2048 x 16B chunks ----
  #pragma unroll
  for (int ch = tid; ch < 2048; ch += 128) {
    int m  = ch >> 7;          // row 0..15
    int kc = ch & 127;         // 16B chunk within row (8 halves each)
    int seq = rowBase + m;
    int b   = seq >> 6;
    int rr  = seq & (S - 1);
    const half_t* src;
    if (kc < 64) {             // x_t portion (K < 512)
      PosV p = map_pos(mode, rr, t, lengths[b]);
      src = pv_h + (((size_t)b * S + p.ip) * S + p.jp) * H + kc * 8;
    } else {                   // h_prev portion (K >= 512)
      src = h_in + (size_t)seq * H + (kc - 64) * 8;
    }
    async_copy16((uint32_t)(uintptr_t)&Ablock[m][kc * 8], src);
  }
  wait_async0();
  __syncthreads();

  v8f accI = {}, accF = {}, accG = {}, accO = {};

  const int nCol = colBaseW + (lane & 15);          // B-fragment output column
  const int kh   = (lane >> 4) * 8;                 // B-fragment K sub-offset
  const int am   = lane & 15;                       // A-fragment row
  const int akb  = (lane >> 4) * 8;                 // A-fragment K sub-offset

  #pragma unroll 4
  for (int ks = 0; ks < 32; ++ks) {
    const half_t* abase = &Ablock[am][ks * 32 + akb];
    v16h afrag = ld_frag(abase, abase + 16);

    #pragma unroll
    for (int g = 0; g < 4; ++g) {
      const half_t* wptr = Wp + (((size_t)g * H + nCol) * KDIM + ks * 32 + kh);
      if (ks + 1 < 32) __builtin_prefetch(wptr + 32, 0, 1);  // next K slab
      v16h bfrag = ld_frag(wptr, wptr + 16);
      if      (g == 0) accI = __builtin_amdgcn_wmma_f32_16x16x32_f16(false, afrag, false, bfrag, (short)0, accI, false, false);
      else if (g == 1) accF = __builtin_amdgcn_wmma_f32_16x16x32_f16(false, afrag, false, bfrag, (short)0, accF, false, false);
      else if (g == 2) accG = __builtin_amdgcn_wmma_f32_16x16x32_f16(false, afrag, false, bfrag, (short)0, accG, false, false);
      else             accO = __builtin_amdgcn_wmma_f32_16x16x32_f16(false, afrag, false, bfrag, (short)0, accO, false, false);
    }
  }

  // Pointwise LSTM cell update. C/D layout: VGPR r, lanes 0-15 -> M=r, N=lane;
  // lanes 16-31 -> M=r+8, N=lane-16.
  #pragma unroll
  for (int r = 0; r < 8; ++r) {
    int M   = r + ((lane >> 4) << 3);
    int seq = rowBase + M;
    int col = colBaseW + (lane & 15);
    int b   = seq >> 6;
    int rr  = seq & (S - 1);
    int len = lengths[b];
    PosV p = map_pos(mode, rr, t, len);

    float iv = sigmoidf(accI[r] + bsum[0 * H + col]);
    float fv = sigmoidf(accF[r] + bsum[1 * H + col]);
    float gv = tanhf  (accG[r] + bsum[2 * H + col]);
    float ov = sigmoidf(accO[r] + bsum[3 * H + col]);

    size_t si = (size_t)seq * H + col;
    float cold = c_state[si];
    float cn = fv * cold + iv * gv;
    float hn = ov * tanhf(cn);

    bool valid = (p.valid != 0);
    c_state[si] = valid ? cn : cold;
    float hprev = (float)h_in[si];
    h_out[si] = (half_t)(valid ? hn : hprev);
    hy[M][col - colBase0] = valid ? hn : 0.0f;
  }
  __syncthreads();

  // Project y-tile (16 x 64) through Wt chunk into pt with atomics.
  for (int e = tid; e < 16 * T; e += 128) {
    int rloc = e / T;
    int tl   = e % T;
    int seq = rowBase + rloc;
    int b   = seq >> 6;
    int rr  = seq & (S - 1);
    PosV p = map_pos(mode, rr, t, lengths[b]);
    if (!p.valid) continue;
    const float* wrow = Wt + (size_t)tl * (2 * H) + dimOff + colBase0;
    float s = 0.0f;
    #pragma unroll 8
    for (int c2 = 0; c2 < 64; ++c2) s += hy[rloc][c2] * wrow[c2];
    atomicAdd(&pt[(((size_t)b * T + tl) * S + p.ip) * S + p.jp], s);
  }
}

// ---------------------------------------------------------------------------
// Epilogue kernels
// ---------------------------------------------------------------------------
__global__ void mask_pt(float* __restrict__ pt, const int* __restrict__ lengths) {
  size_t n = (size_t)B * T * S * S;
  size_t i = (size_t)blockIdx.x * blockDim.x + threadIdx.x;
  size_t stride = (size_t)gridDim.x * blockDim.x;
  for (; i < n; i += stride) {
    int j  = (int)(i & (S - 1));
    int ii = (int)((i >> 6) & (S - 1));
    int t  = (int)((i >> 12) % T);
    int b  = (int)(i / ((size_t)T * S * S));
    bool m = false;
    if (t < T - 1 && ii == j) m = true;
    if (t == T - 1 && j > ii) m = true;
    if (j >= lengths[b])      m = true;
    if (j < ii)               m = true;
    if (m) pt[i] = NEGV;
  }
}

__global__ void lse_kernel(const float* __restrict__ pt, float* __restrict__ labeled) {
  int i = blockIdx.x * blockDim.x + threadIdx.x;
  if (i >= B * S * S) return;
  int j  = i & (S - 1);
  int ii = (i >> 6) & (S - 1);
  int b  = i >> 12;
  float mx = -3.4e38f;
  for (int t = 0; t < T; ++t) {
    float v = pt[(((size_t)b * T + t) * S + ii) * S + j];
    mx = fmaxf(mx, v);
  }
  float s = 0.0f;
  for (int t = 0; t < T; ++t)
    s += __expf(pt[(((size_t)b * T + t) * S + ii) * S + j] - mx);
  labeled[i] = mx + logf(s);
}

__global__ void build_lap(const float* __restrict__ labeled, const int* __restrict__ lengths,
                          float* __restrict__ lap) {
  int b = blockIdx.x;
  int j = threadIdx.x;
  int len = lengths[b];
  const float* L = labeled + (size_t)b * S * S;
  float s = 0.0f;
  for (int i = 0; i < S; ++i) {
    float sc = (i < len && j < len) ? L[i * S + j] : NEGV;
    float A  = (i == j) ? 0.0f : (__expf(sc) + EPSV);
    s += A;
  }
  for (int i = 0; i < S; ++i) {
    float sc = (i < len && j < len) ? L[i * S + j] : NEGV;
    float A  = (i == j) ? 0.0f : (__expf(sc) + EPSV);
    float v  = -A + ((i == j) ? s : 0.0f);
    if (i == 0) {
      float dsc = (j < len) ? L[j * S + j] : NEGV;
      v = __expf(dsc);
    }
    if (i == j && j >= len) v += 1.0f;
    lap[((size_t)b * S + i) * S + j] = v;
  }
}

__global__ void lu_logdet(const float* __restrict__ lap, float* __restrict__ logZ) {
  __shared__ float M[S][S + 1];
  __shared__ float ld;
  __shared__ int piv;
  int b = blockIdx.x;
  int tid = threadIdx.x;
  for (int i = 0; i < S; ++i) M[i][tid] = lap[((size_t)b * S + i) * S + tid];
  if (tid == 0) ld = 0.0f;
  __syncthreads();
  for (int k = 0; k < S; ++k) {
    if (tid == 0) {
      int pp = k; float best = fabsf(M[k][k]);
      for (int i = k + 1; i < S; ++i) {
        float a = fabsf(M[i][k]);
        if (a > best) { best = a; pp = i; }
      }
      piv = pp;
    }
    __syncthreads();
    if (piv != k) { float tmp = M[k][tid]; M[k][tid] = M[piv][tid]; M[piv][tid] = tmp; }
    __syncthreads();
    if (tid == 0) ld += logf(fabsf(M[k][k]) + 1e-30f);
    float pivval = M[k][k];
    int i = tid;
    if (i > k) {
      float l = M[i][k] / pivval;
      for (int j = k; j < S; ++j) M[i][j] -= l * M[k][j];
    }
    __syncthreads();
  }
  if (tid == 0) logZ[b] = ld;
}

__global__ void expand_pe(const float* __restrict__ pt, float* __restrict__ pe) {
  size_t n = (size_t)B * T * (S + 1) * (S + 1);
  size_t i = (size_t)blockIdx.x * blockDim.x + threadIdx.x;
  size_t stride = (size_t)gridDim.x * blockDim.x;
  for (; i < n; i += stride) {
    int rc = (int)(i % ((S + 1) * (S + 1)));
    int bt = (int)(i / ((S + 1) * (S + 1)));
    int t = bt % T;
    int b = bt / T;
    int r  = rc / (S + 1);
    int cl = rc % (S + 1);
    float v;
    if      (r == cl) v = 0.0f;
    else if (cl == 0) v = NEGV;
    else if (r == 0)  v = pt[(((size_t)b * T + t) * S + (cl - 1)) * S + (cl - 1)];
    else              v = pt[(((size_t)b * T + t) * S + (r - 1)) * S + (cl - 1)];
    pe[i] = v;
  }
}

// ---------------------------------------------------------------------------
// Host launcher
// ---------------------------------------------------------------------------
extern "C" void kernel_launch(void* const* d_in, const int* in_sizes, int n_in,
                              void* d_out, int out_size, void* d_ws, size_t ws_size,
                              hipStream_t stream) {
  (void)in_sizes; (void)n_in; (void)out_size; (void)ws_size;

  const int*   lengths = (const int*)  d_in[2];
  const float* sv      = (const float*)d_in[3];
  const float* Wih_f   = (const float*)d_in[5];
  const float* Whh_f   = (const float*)d_in[6];
  const float* bih_f   = (const float*)d_in[7];
  const float* bhh_f   = (const float*)d_in[8];
  const float* Wih_b   = (const float*)d_in[9];
  const float* Whh_b   = (const float*)d_in[10];
  const float* bih_b   = (const float*)d_in[11];
  const float* bhh_b   = (const float*)d_in[12];
  const float* Wt      = (const float*)d_in[13];
  const float* bt      = (const float*)d_in[14];

  char* ws = (char*)d_ws;
  auto alloc = [&](size_t bytes) {
    char* p = ws;
    ws += (bytes + 255) & ~(size_t)255;
    return p;
  };
  half_t* pv_h    = (half_t*)alloc((size_t)B * S * S * H * sizeof(half_t));   // 32 MB
  half_t* Wp      = (half_t*)alloc((size_t)2 * 4 * H * KDIM * sizeof(half_t));// 8 MB
  float*  bsum    = (float*) alloc((size_t)2 * 4 * H * sizeof(float));
  half_t* h0      = (half_t*)alloc((size_t)NSEQ * H * sizeof(half_t));
  half_t* h1      = (half_t*)alloc((size_t)NSEQ * H * sizeof(half_t));
  float*  c_state = (float*) alloc((size_t)NSEQ * H * sizeof(float));
  float*  pt      = (float*) alloc((size_t)B * T * S * S * sizeof(float));
  float*  labeled = (float*) alloc((size_t)B * S * S * sizeof(float));
  float*  lap     = (float*) alloc((size_t)B * S * S * sizeof(float));

  // ---- prep ----
  scatter_pv<<<4096, 256, 0, stream>>>(sv, pv_h);
  pack_w<<<2048, 256, 0, stream>>>(Wih_f, Whh_f, Wih_b, Whh_b, Wp);
  pack_b<<<(2 * 4 * H + 255) / 256, 256, 0, stream>>>(bih_f, bhh_f, bih_b, bhh_b, bsum);
  init_pt<<<1024, 256, 0, stream>>>(pt, bt);

  // ---- 4 LSTM passes x 64 timesteps ----
  // mode: 0=rows_f(uses backward params, dims [0,H))   1=rows_b(fwd params, dims [H,2H))
  //       2=cols_f(fwd params,      dims [0,H))        3=cols_b(bwd params, dims [H,2H))
  const int dirs[4]    = {1, 0, 0, 1};
  const int dimoffs[4] = {0, H, 0, H};
  for (int mode = 0; mode < 4; ++mode) {
    zero_u32<<<512, 256, 0, stream>>>((uint32_t*)h0, (size_t)NSEQ * H * sizeof(half_t) / 4);
    zero_u32<<<512, 256, 0, stream>>>((uint32_t*)c_state, (size_t)NSEQ * H);
    const half_t* Wd = Wp   + (size_t)dirs[mode] * 4 * H * KDIM;
    const float*  bd = bsum + (size_t)dirs[mode] * 4 * H;
    half_t* hin = h0;
    half_t* hout = h1;
    for (int t = 0; t < S; ++t) {
      lstm_step<<<256, 128, 0, stream>>>(pv_h, Wd, bd, hin, hout, c_state,
                                         lengths, Wt, pt, t, mode, dimoffs[mode]);
      half_t* tmp = hin; hin = hout; hout = tmp;
    }
  }

  // ---- epilogue ----
  float* pe   = (float*)d_out;
  float* logZ = pe + (size_t)B * T * (S + 1) * (S + 1);

  mask_pt<<<1024, 256, 0, stream>>>(pt, lengths);
  lse_kernel<<<(B * S * S + 255) / 256, 256, 0, stream>>>(pt, labeled);
  build_lap<<<B, S, 0, stream>>>(labeled, lengths, lap);
  lu_logdet<<<B, S, 0, stream>>>(lap, logZ);
  expand_pe<<<1024, 256, 0, stream>>>(pt, pe);
}
